// GNN_SeasonalityBlock_44178033607256
// MI455X (gfx1250) — matrix operand order repaired and checked by reading, not compiled, not run on hardware
//
#include <hip/hip_runtime.h>

typedef __attribute__((ext_vector_type(16))) _Float16 v16h;
typedef __attribute__((ext_vector_type(8)))  _Float16 v8h;
typedef __attribute__((ext_vector_type(8)))  float    v8f;

#define BN_EPS 1e-5f

// ---------------- pack A: fp32 [M,K] -> f16 [M,Kp] zero-padded ----------------
__global__ void k_pack_a(const float* __restrict__ src, _Float16* __restrict__ dst,
                         int M, int K, int Kp) {
  long long idx = (long long)blockIdx.x * blockDim.x + threadIdx.x;
  long long total = (long long)M * Kp;
  if (idx >= total) return;
  int k = (int)(idx % Kp);
  long long m = idx / Kp;
  float v = (k < K) ? src[m * (long long)K + k] : 0.0f;
  dst[idx] = (_Float16)v;
}

// ---- pack B: fp32 [K,N] row-major -> swizzled f16 WMMA B-operand tiles ------
// dst element index: ((nt*numKt + kt)*32 + lane)*16 + j
//   k = kt*32 + 16*(j>>3) + 8*(lane>>4) + (j&7) ; n = nt*16 + (lane&15)
__global__ void k_pack_b(const float* __restrict__ W, _Float16* __restrict__ dst,
                         int K, int N, int numKt, int ntilesN) {
  long long idx = (long long)blockIdx.x * blockDim.x + threadIdx.x;
  long long total = (long long)ntilesN * numKt * 512;
  if (idx >= total) return;
  int j    = (int)(idx & 15);
  int lane = (int)((idx >> 4) & 31);
  long long t = idx >> 9;
  int kt = (int)(t % numKt);
  int nt = (int)(t / numKt);
  int k = kt * 32 + ((j >> 3) << 4) + ((lane >> 4) << 3) + (j & 7);
  int n = nt * 16 + (lane & 15);
  float v = (k < K && n < N) ? W[(long long)k * N + n] : 0.0f;
  dst[idx] = (_Float16)v;
}

// ------- WMMA GEMM, 16x64 per wave (A fragment reused across 4 N-tiles) -----
// block = 256 threads = 8 waves = 2 M-tiles x 4 N-groups (each group = 4 N-tiles)
// All memory accesses are expressed as base-arg + 64-bit index so the address
// space stays "global" (global_load_b128, no flat/DScnt traffic).
__global__ void k_gemm_wmma(const _Float16* __restrict__ A, const _Float16* __restrict__ Bsw,
                            const float* __restrict__ bias, float* __restrict__ C,
                            int M, int Kp, int Nout, int ldc, int ntilesN, int relu) {
  int lane = threadIdx.x & 31;
  int wave = threadIdx.x >> 5;                 // 0..7
  int mt   = blockIdx.x * 2 + (wave >> 2);
  int ntg  = blockIdx.y * 4 + (wave & 3);      // group of 4 consecutive N-tiles
  int ntgTotal = (ntilesN + 3) >> 2;
  if (ntg >= ntgTotal) return;                 // wave-uniform
  if (mt * 16 >= M) return;                    // wave-uniform
  int numKt = Kp >> 5;
  int half = lane >> 4;
  int mloc = lane & 15;
  int mm   = mt * 16 + mloc;
  int arow = (mm < M) ? mm : (M - 1);
  long long abase = (long long)arow * Kp + 8 * half;

  int nt0 = ntg * 4 + 0, nt1 = ntg * 4 + 1, nt2 = ntg * 4 + 2, nt3 = ntg * 4 + 3;
  int nc0 = (nt0 < ntilesN) ? nt0 : (ntilesN - 1);   // clamp: wasted math, guarded store
  int nc1 = (nt1 < ntilesN) ? nt1 : (ntilesN - 1);
  int nc2 = (nt2 < ntilesN) ? nt2 : (ntilesN - 1);
  int nc3 = (nt3 < ntilesN) ? nt3 : (ntilesN - 1);
  long long lane16 = (long long)lane * 16;
  long long bo0 = (long long)nc0 * numKt * 512 + lane16;
  long long bo1 = (long long)nc1 * numKt * 512 + lane16;
  long long bo2 = (long long)nc2 * numKt * 512 + lane16;
  long long bo3 = (long long)nc3 * numKt * 512 + lane16;

  v8f acc0 = {}, acc1 = {}, acc2 = {}, acc3 = {};
  for (int kt = 0; kt < numKt; ++kt) {
    long long ka = abase + (long long)kt * 32;
    v8h alo = *(const v8h*)(A + ka);
    v8h ahi = *(const v8h*)(A + ka + 16);
    v16h a = __builtin_shufflevector(alo, ahi, 0,1,2,3,4,5,6,7,8,9,10,11,12,13,14,15);
    long long kb = (long long)kt * 512;
    v16h b0 = *(const v16h*)(Bsw + bo0 + kb);
    v16h b1 = *(const v16h*)(Bsw + bo1 + kb);
    v16h b2 = *(const v16h*)(Bsw + bo2 + kb);
    v16h b3 = *(const v16h*)(Bsw + bo3 + kb);
    acc0 = __builtin_amdgcn_wmma_f32_16x16x32_f16(false, a, false, b0, (short)0, acc0, false, false);
    acc1 = __builtin_amdgcn_wmma_f32_16x16x32_f16(false, a, false, b1, (short)0, acc1, false, false);
    acc2 = __builtin_amdgcn_wmma_f32_16x16x32_f16(false, a, false, b2, (short)0, acc2, false, false);
    acc3 = __builtin_amdgcn_wmma_f32_16x16x32_f16(false, a, false, b3, (short)0, acc3, false, false);
  }

  int rowbase = mt * 16 + 8 * half;            // C layout: m = r + 8*(lane>>4)
#define GEMM_EPILOGUE(NT, ACC)                                            \
  if ((NT) < ntilesN) {                                                   \
    int n = (NT) * 16 + mloc;                 /* C layout: n = lane&15 */ \
    if (n < Nout) {                                                       \
      float bv = bias ? bias[n] : 0.0f;                                   \
      _Pragma("unroll")                                                   \
      for (int r = 0; r < 8; ++r) {                                       \
        int row = rowbase + r;                                            \
        if (row < M) {                                                    \
          float v = (ACC)[r] + bv;                                        \
          if (relu) v = fmaxf(v, 0.0f);                                   \
          C[(long long)row * ldc + n] = v;                                \
        }                                                                 \
      }                                                                   \
    }                                                                     \
  }
  GEMM_EPILOGUE(nt0, acc0)
  GEMM_EPILOGUE(nt1, acc1)
  GEMM_EPILOGUE(nt2, acc2)
  GEMM_EPILOGUE(nt3, acc3)
#undef GEMM_EPILOGUE
}

// ---------------- BN stats: per-channel sum / sumsq (N must == blockDim) -----
__global__ void k_bn_stats(const float* __restrict__ y, float* __restrict__ sums,
                           int M, int N) {
  int c = threadIdx.x;
  float s = 0.0f, s2 = 0.0f;
  for (int r = blockIdx.x; r < M; r += gridDim.x) {
    float v = y[(long long)r * N + c];
    s += v; s2 += v * v;
  }
  atomicAdd(&sums[c], s);
  atomicAdd(&sums[N + c], s2);
}

__global__ void k_bn_finalize(const float* __restrict__ sums,
                              const float* __restrict__ gamma, const float* __restrict__ beta,
                              float* __restrict__ scale, float* __restrict__ shift,
                              int M, int N) {
  int c = blockIdx.x * blockDim.x + threadIdx.x;
  if (c >= N) return;
  float inv_m = 1.0f / (float)M;
  float mean = sums[c] * inv_m;
  float var  = sums[N + c] * inv_m - mean * mean;
  float sc   = gamma[c] * rsqrtf(var + BN_EPS);
  scale[c] = sc;
  shift[c] = beta[c] - mean * sc;
}

// apply BN affine and repack as f16 A-operand for next GEMM
__global__ void k_bn_apply(const float* __restrict__ y, const float* __restrict__ scale,
                           const float* __restrict__ shift, _Float16* __restrict__ a16,
                           int M, int N) {
  long long idx = (long long)blockIdx.x * blockDim.x + threadIdx.x;
  long long total = (long long)M * N;
  if (idx >= total) return;
  int c = (int)(idx % N);
  a16[idx] = (_Float16)(y[idx] * scale[c] + shift[c]);
}

// ---------------- degree / dinv ----------------------------------------------
__global__ void k_deg_init(float* __restrict__ deg, int n) {
  int i = blockIdx.x * blockDim.x + threadIdx.x;
  if (i < n) deg[i] = 1.0f;               // self-loop
}
__global__ void k_deg_accum(const int* __restrict__ dstE, float* __restrict__ deg, int E) {
  for (int e = blockIdx.x * blockDim.x + threadIdx.x; e < E; e += gridDim.x * blockDim.x)
    atomicAdd(&deg[dstE[e]], 1.0f);
}
__global__ void k_rsqrt_inplace(float* __restrict__ a, int n) {
  int i = blockIdx.x * blockDim.x + threadIdx.x;
  if (i < n) a[i] = rsqrtf(a[i]);
}

// ---------------- GCN scatter: agg[dst] += xw[src] * dinv[src]*dinv[dst] -----
__global__ void k_gcn_scatter(const float* __restrict__ xw, const int* __restrict__ srcE,
                              const int* __restrict__ dstE, const float* __restrict__ dinv,
                              float* __restrict__ agg, int E, int N) {
  int c = threadIdx.x;                    // channel, blockDim == N
  for (int e = blockIdx.x; e < E; e += gridDim.x) {
    int s = srcE[e], d = dstE[e];
    float w = dinv[s] * dinv[d];
    atomicAdd(&agg[(long long)d * N + c], xw[(long long)s * N + c] * w);
  }
}

// self-loop + bias + relu (writes in-place over xw buffer)
__global__ void k_gcn_post(const float* __restrict__ agg, float* __restrict__ xw,
                           const float* __restrict__ dinv, const float* __restrict__ bias,
                           int M, int N) {
  long long idx = (long long)blockIdx.x * blockDim.x + threadIdx.x;
  long long total = (long long)M * N;
  if (idx >= total) return;
  int c = (int)(idx % N);
  int row = (int)(idx / N);
  float di = dinv[row];
  float v = agg[idx] + xw[idx] * di * di + bias[c];
  xw[idx] = fmaxf(v, 0.0f);
}

// ---------------- tiny fp32 gemm for Wb = tbW @ basis (256x168 / 256x24) ------
__global__ void k_small_gemm(const float* __restrict__ A, const float* __restrict__ B,
                             float* __restrict__ C, int M, int K, int N) {
  long long idx = (long long)blockIdx.x * blockDim.x + threadIdx.x;
  if (idx >= (long long)M * N) return;
  int j = (int)(idx % N);
  int i = (int)(idx / N);
  float s = 0.0f;
  for (int k = 0; k < K; ++k) s += A[(long long)i * K + k] * B[(long long)k * N + j];
  C[idx] = s;
}

// =============================================================================
extern "C" void kernel_launch(void* const* d_in, const int* in_sizes, int n_in,
                              void* d_out, int out_size, void* d_ws, size_t ws_size,
                              hipStream_t stream) {
  const int LB = 168, LF = 24, THB = 167, THF = 23, H = 256;
  const float* x  = (const float*)d_in[0];
  const int* ei   = (const int*)d_in[1];
  const float* w1 = (const float*)d_in[2];  const float* b1  = (const float*)d_in[3];
  const float* g1 = (const float*)d_in[4];  const float* bt1 = (const float*)d_in[5];
  const float* w2 = (const float*)d_in[6];  const float* b2  = (const float*)d_in[7];
  const float* g2 = (const float*)d_in[8];  const float* bt2 = (const float*)d_in[9];
  const float* gw1 = (const float*)d_in[10]; const float* gb1  = (const float*)d_in[11];
  const float* gg1 = (const float*)d_in[12]; const float* gbt1 = (const float*)d_in[13];
  const float* gw2 = (const float*)d_in[14]; const float* gb2  = (const float*)d_in[15];
  const float* gg2 = (const float*)d_in[16]; const float* gbt2 = (const float*)d_in[17];
  const float* tbW = (const float*)d_in[18]; const float* tfW  = (const float*)d_in[19];
  const float* basis_b = (const float*)d_in[20]; const float* basis_f = (const float*)d_in[21];

  const int Nn = in_sizes[0] / LB;        // 50000
  const int E  = in_sizes[1] / 2;         // 1600000
  const int* srcE = ei;
  const int* dstE = ei + E;

  const int Kp1 = 192;                    // 168 padded to 32-mult
  const int NpB = 176;                    // 168 padded to 16-mult
  const int NpF = 32;                     // 24  padded to 16-mult

  // ----- workspace carve-up (256B aligned) -----
  size_t off = 0;
  auto carve = [&](size_t bytes) -> char* {
    char* p = (char*)d_ws + off;
    off += (bytes + 255) & ~(size_t)255;
    return p;
  };
  _Float16* a16  = (_Float16*)carve((size_t)Nn * H * sizeof(_Float16));
  float*    y32  = (float*)carve((size_t)Nn * H * sizeof(float));
  float*    agg  = (float*)carve((size_t)Nn * H * sizeof(float));
  float*    dinv = (float*)carve((size_t)Nn * sizeof(float));
  float*    sums = (float*)carve(2 * H * sizeof(float));
  float*    scl  = (float*)carve(H * sizeof(float));
  float*    shf  = (float*)carve(H * sizeof(float));
  _Float16* bw1  = (_Float16*)carve((size_t)Kp1 * H * sizeof(_Float16));
  _Float16* bw2  = (_Float16*)carve((size_t)H * H * sizeof(_Float16));
  _Float16* bgw1 = (_Float16*)carve((size_t)H * H * sizeof(_Float16));
  _Float16* bgw2 = (_Float16*)carve((size_t)H * H * sizeof(_Float16));
  _Float16* bWb  = (_Float16*)carve((size_t)H * NpB * sizeof(_Float16));
  _Float16* bWf  = (_Float16*)carve((size_t)H * NpF * sizeof(_Float16));
  float*    Wb32 = (float*)carve((size_t)H * LB * sizeof(float));
  float*    Wf32 = (float*)carve((size_t)H * LF * sizeof(float));
  (void)ws_size;

  auto blocks = [](long long total, int bs) { return (int)((total + bs - 1) / bs); };
  const int BS = 256;

  // ----- one-time-per-launch prep (deterministic) -----
  k_pack_a<<<blocks((long long)Nn * Kp1, BS), BS, 0, stream>>>(x, a16, Nn, LB, Kp1);
  k_pack_b<<<blocks((long long)(H/16) * (Kp1/32) * 512, BS), BS, 0, stream>>>(w1,  bw1,  LB, H, Kp1/32, H/16);
  k_pack_b<<<blocks((long long)(H/16) * (H/32) * 512,  BS), BS, 0, stream>>>(w2,  bw2,  H,  H, H/32,  H/16);
  k_pack_b<<<blocks((long long)(H/16) * (H/32) * 512,  BS), BS, 0, stream>>>(gw1, bgw1, H,  H, H/32,  H/16);
  k_pack_b<<<blocks((long long)(H/16) * (H/32) * 512,  BS), BS, 0, stream>>>(gw2, bgw2, H,  H, H/32,  H/16);
  // fold basis into theta: Wb = tbW @ basis_b ; Wf = tfW @ basis_f
  k_small_gemm<<<blocks((long long)H * LB, BS), BS, 0, stream>>>(tbW, basis_b, Wb32, H, THB, LB);
  k_small_gemm<<<blocks((long long)H * LF, BS), BS, 0, stream>>>(tfW, basis_f, Wf32, H, THF, LF);
  k_pack_b<<<blocks((long long)(NpB/16) * (H/32) * 512, BS), BS, 0, stream>>>(Wb32, bWb, H, LB, H/32, NpB/16);
  k_pack_b<<<blocks((long long)(NpF/16) * (H/32) * 512, BS), BS, 0, stream>>>(Wf32, bWf, H, LF, H/32, NpF/16);

  // degrees -> dinv
  k_deg_init<<<blocks(Nn, BS), BS, 0, stream>>>(dinv, Nn);
  k_deg_accum<<<2048, BS, 0, stream>>>(dstE, dinv, E);
  k_rsqrt_inplace<<<blocks(Nn, BS), BS, 0, stream>>>(dinv, Nn);

  const int Mt  = (Nn + 15) / 16;         // 3125
  const int Mb  = (Mt + 1) / 2;           // 2 M-tiles per block
  dim3 gH(Mb, 1);                         // H: 16 N-tiles = 4 groups -> one block-row
  dim3 gB(Mb, 1);                         // backcast: 11 N-tiles = 3 groups
  dim3 gF(Mb, 1);                         // forecast: 2 N-tiles = 1 group

  auto bn_block = [&](const float* gamma, const float* beta) {
    hipMemsetAsync(sums, 0, 2 * H * sizeof(float), stream);
    k_bn_stats<<<512, H, 0, stream>>>(y32, sums, Nn, H);
    k_bn_finalize<<<1, H, 0, stream>>>(sums, gamma, beta, scl, shf, Nn, H);
    k_bn_apply<<<blocks((long long)Nn * H, BS), BS, 0, stream>>>(y32, scl, shf, a16, Nn, H);
  };

  // ----- MLP layer 1: y = relu(x @ w1 + b1); BN -> a16 -----
  k_gemm_wmma<<<gH, 256, 0, stream>>>(a16, bw1, b1, y32, Nn, Kp1, H, H, H/16, 1);
  bn_block(g1, bt1);
  // ----- MLP layer 2 -----
  k_gemm_wmma<<<gH, 256, 0, stream>>>(a16, bw2, b2, y32, Nn, H, H, H, H/16, 1);
  bn_block(g2, bt2);

  // ----- GCN layer 1 -----
  k_gemm_wmma<<<gH, 256, 0, stream>>>(a16, bgw1, nullptr, y32, Nn, H, H, H, H/16, 0);
  hipMemsetAsync(agg, 0, (size_t)Nn * H * sizeof(float), stream);
  k_gcn_scatter<<<8192, H, 0, stream>>>(y32, srcE, dstE, dinv, agg, E, H);
  k_gcn_post<<<blocks((long long)Nn * H, BS), BS, 0, stream>>>(agg, y32, dinv, gb1, Nn, H);
  bn_block(gg1, gbt1);

  // ----- GCN layer 2 -----
  k_gemm_wmma<<<gH, 256, 0, stream>>>(a16, bgw2, nullptr, y32, Nn, H, H, H, H/16, 0);
  hipMemsetAsync(agg, 0, (size_t)Nn * H * sizeof(float), stream);
  k_gcn_scatter<<<8192, H, 0, stream>>>(y32, srcE, dstE, dinv, agg, E, H);
  k_gcn_post<<<blocks((long long)Nn * H, BS), BS, 0, stream>>>(agg, y32, dinv, gb2, Nn, H);
  bn_block(gg2, gbt2);

  // ----- heads: backcast = h @ Wb ; forecast = h @ Wf (directly into d_out) --
  float* out_back = (float*)d_out;
  float* out_fore = out_back + (long long)Nn * LB;
  k_gemm_wmma<<<gB, 256, 0, stream>>>(a16, bWb, nullptr, out_back, Nn, H, LB, LB, NpB/16, 0);
  k_gemm_wmma<<<gF, 256, 0, stream>>>(a16, bWf, nullptr, out_fore, Nn, H, LF, LF, NpF/16, 0);
}
